// GraphEncoder_7524782702966
// MI455X (gfx1250) — compile-verified, matrix operand order checked by reading, not run on hardware
//
#include <hip/hip_runtime.h>
#include <hip/hip_bf16.h>
#include <cstdint>

typedef __attribute__((ext_vector_type(16))) _Float16 v16h;
typedef __attribute__((ext_vector_type(8)))  float    v8f;

#define NEGS 0.2f

// ---------------------------------------------------------------------------
// utility: fill a float buffer with a constant
// ---------------------------------------------------------------------------
__global__ void fill_f32_kernel(float* __restrict__ p, float v, long long n) {
    long long i = (long long)blockIdx.x * blockDim.x + threadIdx.x;
    if (i < n) p[i] = v;
}

// ---------------------------------------------------------------------------
// self-loop edge_attr mean: accumulate deg and per-dst edge_attr sums
// ---------------------------------------------------------------------------
__global__ void selfloop_accum_kernel(const long long* __restrict__ ei,
                                      const float* __restrict__ eattr,
                                      float* __restrict__ deg,
                                      float* __restrict__ easum, int E) {
    int e = blockIdx.x * blockDim.x + threadIdx.x;
    if (e >= E) return;
    int dst = (int)ei[E + e];
    atomicAdd(&deg[dst], 1.0f);
#pragma unroll
    for (int j = 0; j < 4; ++j)
        atomicAdd(&easum[dst * 4 + j], eattr[e * 4 + j]);
}

// Build concatenated edge_attr array: first E rows copied, then N self-loop means
__global__ void build_ea_kernel(const float* __restrict__ eattr,
                                const float* __restrict__ deg,
                                const float* __restrict__ easum,
                                float* __restrict__ ea, int E, int N) {
    long long i = (long long)blockIdx.x * blockDim.x + threadIdx.x;
    long long tot = (long long)(E + N) * 4;
    if (i >= tot) return;
    if (i < (long long)E * 4) {
        ea[i] = eattr[i];
    } else {
        long long r = i - (long long)E * 4;
        int node = (int)(r >> 2);
        int j = (int)(r & 3);
        ea[i] = easum[node * 4 + j] / fmaxf(deg[node], 1.0f);
    }
}

// ---------------------------------------------------------------------------
// Pack Wl/Wr [din x 128] (f32) into per-lane WMMA B-fragment order (f16):
// Bp[(((side*ntiles + t)*8 + w)*32 + lane)*16 + h] = W[t*32 + kmap(h,g)][w*16 + lane%16]
// with kmap(h,g) = (h&7) + 8g + 16*(h>>3)  (ISA 16-bit B layout, g = lane/16).
// ---------------------------------------------------------------------------
__global__ void pack_w_kernel(const float* __restrict__ Wl,
                              const float* __restrict__ Wr,
                              _Float16* __restrict__ Bp, int din, int ntiles) {
    int idx = blockIdx.x * blockDim.x + threadIdx.x;
    int total = 2 * ntiles * 8 * 32;
    if (idx >= total) return;
    int lane = idx & 31;
    int w    = (idx >> 5) & 7;
    int t    = (idx >> 8) % ntiles;
    int side = (idx >> 8) / ntiles;
    const float* W = side ? Wr : Wl;
    int g    = lane >> 4;
    int ncol = w * 16 + (lane & 15);
    _Float16* dst = Bp + (size_t)idx * 16;
#pragma unroll
    for (int h = 0; h < 16; ++h) {
        int k = t * 32 + (h & 7) + (g << 3) + ((h >> 3) << 4);
        dst[h] = (k < din) ? (_Float16)W[(size_t)k * 128 + ncol] : (_Float16)0.0f;
    }
}

// ---------------------------------------------------------------------------
// Node GEMM with WMMA: out = X[N x DIN] @ W[DIN x 128] + b.
// DIN is a template parameter -> no tail branches, compile-time trip count,
// full unroll. No LDS: each lane loads its two 8-float A-row runs as aligned
// float4s; B comes pre-packed as one contiguous v16h per lane per k-tile.
// grid.y selects (bl,xl)/(br,xr).
// block = 256 threads = 8 waves; wave w owns output column tile w (16 cols).
// ---------------------------------------------------------------------------
template <int DIN>
__global__ void node_gemm_wmma_kernel(const float* __restrict__ X,
                                      const _Float16* __restrict__ Bp,
                                      const float* __restrict__ bl,
                                      const float* __restrict__ br,
                                      float* __restrict__ xl,
                                      float* __restrict__ xr, int n) {
    constexpr int NT = (DIN + 31) / 32;      // k-tiles

    const int   side = blockIdx.y;
    const float* bias = side ? br : bl;
    float*       out  = side ? xr : xl;

    const int row0 = blockIdx.x * 16;
    const int wave = threadIdx.x >> 5;
    const int lane = threadIdx.x & 31;
    const int g    = lane >> 4;
    const int m    = lane & 15;
    const int ncol = wave * 16 + m;

    int rr = row0 + m;
    if (rr >= n) rr = n - 1;                 // clamp for loads; stores guarded
    const float* Xr  = X + (size_t)rr * DIN;
    const v16h*  Bpv = (const v16h*)Bp + ((size_t)side * NT * 8 + wave) * 32 + lane;

    v8f acc = {};
#pragma unroll
    for (int t = 0; t < NT; ++t) {
        const int kA = t * 32 + (g << 3);    // first 8-float run (always < DIN)
        const int kB = kA + 16;              // second 8-float run
        v16h a;
        {
            const float4* p = (const float4*)(Xr + kA);
            float4 r0 = p[0], r1 = p[1];
            a[0] = (_Float16)r0.x; a[1] = (_Float16)r0.y;
            a[2] = (_Float16)r0.z; a[3] = (_Float16)r0.w;
            a[4] = (_Float16)r1.x; a[5] = (_Float16)r1.y;
            a[6] = (_Float16)r1.z; a[7] = (_Float16)r1.w;
        }
        if constexpr (true) {
            if (kB + 7 < DIN) {              // compile-time decidable per tile
                const float4* q = (const float4*)(Xr + kB);
                float4 r2 = q[0], r3 = q[1];
                a[8]  = (_Float16)r2.x; a[9]  = (_Float16)r2.y;
                a[10] = (_Float16)r2.z; a[11] = (_Float16)r2.w;
                a[12] = (_Float16)r3.x; a[13] = (_Float16)r3.y;
                a[14] = (_Float16)r3.z; a[15] = (_Float16)r3.w;
            } else {
                a[8]  = (_Float16)0.0f; a[9]  = (_Float16)0.0f;
                a[10] = (_Float16)0.0f; a[11] = (_Float16)0.0f;
                a[12] = (_Float16)0.0f; a[13] = (_Float16)0.0f;
                a[14] = (_Float16)0.0f; a[15] = (_Float16)0.0f;
            }
        }
        v16h b = Bpv[(size_t)t * 8 * 32];    // packed: stride 8*32 fragments per tile
        acc = __builtin_amdgcn_wmma_f32_16x16x32_f16(
            /*neg_a=*/false, a, /*neg_b=*/false, b,
            /*c_mod=*/(short)0, acc, /*reuse_a=*/false, /*reuse_b=*/false);
    }

    const float bv = bias[ncol];
#pragma unroll
    for (int r = 0; r < 8; ++r) {
        int mm = r + (g << 3);               // ISA C/D layout: row = r + 8*g
        int ro = row0 + mm;
        if (ro < n) out[(size_t)ro * 128 + ncol] = acc[r] + bv;
    }
}

// NOTE on kB guard: kA = t*32 + 8g with g in {0,1}. For DIN=128 every kB run is
// in-bounds for both g; for DIN=16 it is out-of-bounds for both g. The
// condition (kB+7 < DIN) is therefore invariant across the wave AND across g
// only when DIN is a multiple of 32 or DIN<=16; both instantiations used here
// (16, 128) satisfy that, so the compiler folds... (kB depends on g, but
// t*32+8g+16+7 < DIN reduces to t*32+31 < DIN for g=1 and t*32+23 < DIN for
// g=0 — for DIN=128 both true for all t<4; for DIN=16 both false. The branch
// is thus uniform and constant-foldable after inlining g's range? g is not a
// compile-time constant, so the compiler keeps a compare — but for DIN=128,
// kB+7 = t*32+8g+23 <= 127 for all g, and LLVM's range analysis on (lane>>4)
// proves it; for DIN=16 it proves the negation. Either way no per-iteration
// divergence survives in practice for these instantiations.)

// ---------------------------------------------------------------------------
// float atomic max via monotonic int mapping
// ---------------------------------------------------------------------------
__device__ __forceinline__ void atomicMaxFloat(float* addr, float val) {
    if (val >= 0.0f)
        atomicMax((int*)addr, __float_as_int(val));
    else
        atomicMin((unsigned int*)addr, (unsigned int)__float_as_int(val));
}

// ---------------------------------------------------------------------------
// Edge logits: one wave per edge; lane l owns channel l of each head.
// ee computed on the fly (K=4). Writes logit[e][h] and atomic-max lmax[dst][h].
// ---------------------------------------------------------------------------
__global__ void edge_logits_kernel(const long long* __restrict__ ei,
                                   const float4* __restrict__ ea4,
                                   const float* __restrict__ We,
                                   const float* __restrict__ att,
                                   const float* __restrict__ xl,
                                   const float* __restrict__ xr,
                                   float* __restrict__ logit,
                                   float* __restrict__ lmax,
                                   int E, int E2) {
    int e = blockIdx.x * (blockDim.x >> 5) + (threadIdx.x >> 5);
    if (e >= E2) return;
    int lane = threadIdx.x & 31;
    int src, dst;
    if (e < E) { src = (int)ei[e]; dst = (int)ei[E + e]; }
    else       { src = dst = e - E; }

    const float4 eav = ea4[e];

    float lg[4];
#pragma unroll
    for (int h = 0; h < 4; ++h) {
        int ch = h * 32 + lane;
        float ee = eav.x * We[ch] + eav.y * We[128 + ch] +
                   eav.z * We[256 + ch] + eav.w * We[384 + ch];
        float mval = xl[(size_t)src * 128 + ch] + xr[(size_t)dst * 128 + ch] + ee;
        float s = (mval >= 0.0f) ? mval : NEGS * mval;
        lg[h] = s * att[h * 32 + lane];
    }
#pragma unroll
    for (int off = 16; off; off >>= 1) {
        lg[0] += __shfl_xor(lg[0], off, 32);
        lg[1] += __shfl_xor(lg[1], off, 32);
        lg[2] += __shfl_xor(lg[2], off, 32);
        lg[3] += __shfl_xor(lg[3], off, 32);
    }
    if (lane == 0) {
#pragma unroll
        for (int h = 0; h < 4; ++h) {
            logit[(size_t)e * 4 + h] = lg[h];
            atomicMaxFloat(&lmax[dst * 4 + h], lg[h]);
        }
    }
}

// ---------------------------------------------------------------------------
// exp(logit - lmax) in place, accumulate softmax denominators
// ---------------------------------------------------------------------------
__global__ void edge_exp_kernel(const long long* __restrict__ ei,
                                float* __restrict__ logit,
                                const float* __restrict__ lmax,
                                float* __restrict__ den, int E, int E2) {
    long long i = (long long)blockIdx.x * blockDim.x + threadIdx.x;
    if (i >= (long long)E2 * 4) return;
    int e = (int)(i >> 2);
    int h = (int)(i & 3);
    int dst = (e < E) ? (int)ei[E + e] : e - E;
    float lm = lmax[dst * 4 + h];
    if (!isfinite(lm)) lm = 0.0f;
    float ex = expf(logit[i] - lm);
    logit[i] = ex;
    atomicAdd(&den[dst * 4 + h], ex);
}

// ---------------------------------------------------------------------------
// Weighted aggregation: out[dst] += alpha * xl[src]; one wave per edge.
// ---------------------------------------------------------------------------
__global__ void edge_aggregate_kernel(const long long* __restrict__ ei,
                                      const float* __restrict__ ex,
                                      const float* __restrict__ den,
                                      const float* __restrict__ xl,
                                      float* __restrict__ out, int E, int E2) {
    int e = blockIdx.x * (blockDim.x >> 5) + (threadIdx.x >> 5);
    if (e >= E2) return;
    int lane = threadIdx.x & 31;
    int src, dst;
    if (e < E) { src = (int)ei[e]; dst = (int)ei[E + e]; }
    else       { src = dst = e - E; }
#pragma unroll
    for (int h = 0; h < 4; ++h) {
        float alpha = ex[(size_t)e * 4 + h] / (den[dst * 4 + h] + 1e-16f);
        int ch = h * 32 + lane;
        atomicAdd(&out[(size_t)dst * 128 + ch],
                  alpha * xl[(size_t)src * 128 + ch]);
    }
}

// ---------------------------------------------------------------------------
// out += bias, optional ReLU
// ---------------------------------------------------------------------------
__global__ void bias_relu_kernel(float* __restrict__ h, const float* __restrict__ bias,
                                 long long n, int dorelu) {
    long long i = (long long)blockIdx.x * blockDim.x + threadIdx.x;
    if (i >= n * 128) return;
    float v = h[i] + bias[i & 127];
    if (dorelu) v = fmaxf(v, 0.0f);
    h[i] = v;
}

// ---------------------------------------------------------------------------
// global mean pool
// ---------------------------------------------------------------------------
__global__ void pool_count_kernel(const long long* __restrict__ batch,
                                  float* __restrict__ cnt, int n) {
    int i = blockIdx.x * blockDim.x + threadIdx.x;
    if (i >= n) return;
    atomicAdd(&cnt[(int)batch[i]], 1.0f);
}

__global__ void pool_accum_kernel(const long long* __restrict__ batch,
                                  const float* __restrict__ h,
                                  float* __restrict__ pooled, int n) {
    long long i = (long long)blockIdx.x * blockDim.x + threadIdx.x;
    if (i >= (long long)n * 128) return;
    int node = (int)(i >> 7);
    int c = (int)(i & 127);
    atomicAdd(&pooled[(int)batch[node] * 128 + c], h[i]);
}

__global__ void pool_div_kernel(const float* __restrict__ pooled,
                                const float* __restrict__ cnt,
                                float* __restrict__ out, int G) {
    int i = blockIdx.x * blockDim.x + threadIdx.x;
    if (i >= G * 128) return;
    out[i] = pooled[i] / fmaxf(cnt[i >> 7], 1.0f);
}

// ===========================================================================
extern "C" void kernel_launch(void* const* d_in, const int* in_sizes, int n_in,
                              void* d_out, int out_size, void* d_ws, size_t ws_size,
                              hipStream_t stream) {
    (void)n_in; (void)out_size; (void)ws_size;
    const float*     x     = (const float*)d_in[0];
    const float*     eattr = (const float*)d_in[1];
    const long long* ei    = (const long long*)d_in[2];
    const long long* batch = (const long long*)d_in[3];

    const int N  = in_sizes[0] / 16;
    const int E  = in_sizes[1] / 4;
    const int E2 = E + N;
    const int G  = 64;

    const float *Wl[3], *bl[3], *Wr[3], *br[3], *We[3], *att[3], *bias[3];
    for (int l = 0; l < 3; ++l) {
        int b0 = 4 + l * 7;
        Wl[l]   = (const float*)d_in[b0 + 0];
        bl[l]   = (const float*)d_in[b0 + 1];
        Wr[l]   = (const float*)d_in[b0 + 2];
        br[l]   = (const float*)d_in[b0 + 3];
        We[l]   = (const float*)d_in[b0 + 4];
        att[l]  = (const float*)d_in[b0 + 5];
        bias[l] = (const float*)d_in[b0 + 6];
    }

    // ---- carve workspace ----
    char* w = (char*)d_ws;
    auto carve = [&](size_t bytes) -> float* {
        float* p = (float*)w;
        w += (bytes + 255) & ~(size_t)255;
        return p;
    };
    float*     ea     = carve((size_t)E2 * 4 * 4);   // concatenated edge_attr
    float*     deg    = carve((size_t)N * 4);
    float*     easum  = carve((size_t)N * 4 * 4);
    float*     xlb    = carve((size_t)N * 128 * 4);
    float*     xrb    = carve((size_t)N * 128 * 4);
    float*     hA     = carve((size_t)N * 128 * 4);
    float*     hB     = carve((size_t)N * 128 * 4);
    float*     logit  = carve((size_t)E2 * 4 * 4);   // logits, then exp in place
    float*     lmax   = carve((size_t)N * 4 * 4);
    float*     den    = carve((size_t)N * 4 * 4);
    float*     pooled = carve((size_t)G * 128 * 4);
    float*     cnt    = carve((size_t)G * 4);
    _Float16*  Bpack  = (_Float16*)carve(2 * 4 * 8 * 32 * 16 * sizeof(_Float16)); // 64KB max

    auto fill = [&](float* p, float v, long long n) {
        int blocks = (int)((n + 255) / 256);
        hipLaunchKernelGGL(fill_f32_kernel, dim3(blocks), dim3(256), 0, stream, p, v, n);
    };

    // ---- self-loop mean edge_attr (once; reused by all 3 layers) ----
    fill(deg, 0.0f, N);
    fill(easum, 0.0f, (long long)N * 4);
    hipLaunchKernelGGL(selfloop_accum_kernel, dim3((E + 255) / 256), dim3(256), 0, stream,
                       ei, eattr, deg, easum, E);
    hipLaunchKernelGGL(build_ea_kernel, dim3((int)(((long long)E2 * 4 + 255) / 256)),
                       dim3(256), 0, stream, eattr, deg, easum, ea, E, N);

    const int edgeBlocks = (E2 + 7) / 8;   // 8 waves (edges) per 256-thread block

    for (int l = 0; l < 3; ++l) {
        const float* in  = (l == 0) ? x : ((l == 1) ? hA : hB);
        float*       out = (l == 0) ? hA : ((l == 1) ? hB : hA);
        const int din = (l == 0) ? 16 : 128;
        const int ntiles = (din + 31) / 32;

        // pack Wl/Wr into WMMA B-fragment order (f16)
        int packThreads = 2 * ntiles * 8 * 32;
        hipLaunchKernelGGL(pack_w_kernel, dim3((packThreads + 255) / 256), dim3(256), 0,
                           stream, Wl[l], Wr[l], Bpack, din, ntiles);

        // dense node transforms (WMMA), DIN specialized at compile time
        if (l == 0)
            hipLaunchKernelGGL((node_gemm_wmma_kernel<16>), dim3((N + 15) / 16, 2),
                               dim3(256), 0, stream, in, Bpack, bl[l], br[l], xlb, xrb, N);
        else
            hipLaunchKernelGGL((node_gemm_wmma_kernel<128>), dim3((N + 15) / 16, 2),
                               dim3(256), 0, stream, in, Bpack, bl[l], br[l], xlb, xrb, N);

        // attention
        fill(lmax, -INFINITY, (long long)N * 4);
        fill(den, 0.0f, (long long)N * 4);
        hipLaunchKernelGGL(edge_logits_kernel, dim3(edgeBlocks), dim3(256), 0, stream,
                           ei, (const float4*)ea, We[l], att[l], xlb, xrb, logit, lmax, E, E2);
        hipLaunchKernelGGL(edge_exp_kernel, dim3((int)(((long long)E2 * 4 + 255) / 256)),
                           dim3(256), 0, stream, ei, logit, lmax, den, E, E2);

        // aggregate
        fill(out, 0.0f, (long long)N * 128);
        hipLaunchKernelGGL(edge_aggregate_kernel, dim3(edgeBlocks), dim3(256), 0, stream,
                           ei, logit, den, xlb, out, E, E2);
        hipLaunchKernelGGL(bias_relu_kernel, dim3((int)(((long long)N * 128 + 255) / 256)),
                           dim3(256), 0, stream, out, bias[l], (long long)N, (l < 2) ? 1 : 0);
    }

    // ---- global mean pool (final features live in hA) ----
    fill(pooled, 0.0f, (long long)G * 128);
    fill(cnt, 0.0f, G);
    hipLaunchKernelGGL(pool_count_kernel, dim3((N + 255) / 256), dim3(256), 0, stream,
                       batch, cnt, N);
    hipLaunchKernelGGL(pool_accum_kernel, dim3((int)(((long long)N * 128 + 255) / 256)),
                       dim3(256), 0, stream, batch, hA, pooled, N);
    hipLaunchKernelGGL(pool_div_kernel, dim3((G * 128 + 255) / 256), dim3(256), 0, stream,
                       pooled, cnt, (float*)d_out, G);
}
